// GraphormerAttentionHead_8727373545994
// MI455X (gfx1250) — matrix operand order repaired and checked by reading, not compiled, not run on hardware
//
#include <hip/hip_runtime.h>

#define NN    8192
#define DIN   256
#define DH    64
#define CHUNK 32

typedef __attribute__((ext_vector_type(16))) _Float16 v16h;
typedef __attribute__((ext_vector_type(8)))  _Float16 v8h;
typedef __attribute__((ext_vector_type(8)))  float    v8f;
typedef __attribute__((ext_vector_type(8)))  int      v8i;

static __device__ __forceinline__ v8f wmma16(v16h a, v16h b, v8f c) {
  // D = A(16x32 f16) * B(32x16 f16) + C(16x16 f32)
  return __builtin_amdgcn_wmma_f32_16x16x32_f16(false, a, false, b, (short)0, c,
                                                false, false);
}
static __device__ __forceinline__ v16h cat8(v8h lo, v8h hi) {
  return __builtin_shufflevector(lo, hi, 0, 1, 2, 3, 4, 5, 6, 7,
                                 8, 9, 10, 11, 12, 13, 14, 15);
}
static __device__ __forceinline__ v8h shfl16_v8h(v8h x) {
  union { v8h h; int i[4]; } u;
  u.h = x;
#pragma unroll
  for (int k = 0; k < 4; ++k) u.i[k] = __shfl_xor(u.i[k], 16);
  return u.h;
}

// ---------------------------------------------------------------------------
// Phase 1: q = (query Wq^T + bq)/8 -> f16 [N][64];  k -> f16 [N][64]
//          vT -> f16 [64][N];  ptr(int64) -> int32
// ---------------------------------------------------------------------------
__global__ __launch_bounds__(256) void proj_kernel(
    const float* __restrict__ query, const float* __restrict__ key,
    const float* __restrict__ value,
    const float* __restrict__ Wq, const float* __restrict__ bq,
    const float* __restrict__ Wk, const float* __restrict__ bk,
    const float* __restrict__ Wv, const float* __restrict__ bv,
    const long long* __restrict__ ptr,
    _Float16* __restrict__ qh, _Float16* __restrict__ kh,
    _Float16* __restrict__ vT, int* __restrict__ ptr32) {
  int gid  = blockIdx.x * blockDim.x + threadIdx.x;
  int node = gid >> 6;
  int feat = gid & 63;
  if (node >= NN) return;

  const float4* xq = (const float4*)(query + node * DIN);
  const float4* xk = (const float4*)(key   + node * DIN);
  const float4* xv = (const float4*)(value + node * DIN);
  const float4* wq = (const float4*)(Wq + feat * DIN);
  const float4* wk = (const float4*)(Wk + feat * DIN);
  const float4* wv = (const float4*)(Wv + feat * DIN);

  float aq = bq[feat], ak = bk[feat], av = bv[feat];
#pragma unroll 8
  for (int c = 0; c < DIN / 4; ++c) {
    float4 a, w;
    a = xq[c]; w = wq[c];
    aq += a.x * w.x + a.y * w.y + a.z * w.z + a.w * w.w;
    a = xk[c]; w = wk[c];
    ak += a.x * w.x + a.y * w.y + a.z * w.z + a.w * w.w;
    a = xv[c]; w = wv[c];
    av += a.x * w.x + a.y * w.y + a.z * w.z + a.w * w.w;
  }
  qh[node * DH + feat]         = (_Float16)(aq * 0.125f);  // fold 1/sqrt(64)
  kh[node * DH + feat]         = (_Float16)ak;
  vT[(size_t)feat * NN + node] = (_Float16)av;
  if (feat == 0) ptr32[node] = (int)ptr[node];
}

// ---------------------------------------------------------------------------
// Phase 2: flash attention over a j-range, transposed-score orientation.
// One wave32 per 16-query block; lane owns one query column (duplicated in
// the two half-waves). Writes unnormalized partial O plus per-query (M, L).
// ---------------------------------------------------------------------------
__global__ __launch_bounds__(32) void attn_kernel(
    const float* __restrict__ bmat, const _Float16* __restrict__ qh,
    const _Float16* __restrict__ kh, const _Float16* __restrict__ vT,
    const int* __restrict__ ptr32, float* __restrict__ pO,
    float* __restrict__ pM, float* __restrict__ pL, int jlen) {
  const int lane = threadIdx.x;
  const int half = lane >> 4;  // which 16-lane half
  const int lm   = lane & 15;
  const int q0   = blockIdx.x * 16;
  const int kb   = half * 8;            // A-layout K sub-base for this half
  const int js   = blockIdx.y * jlen;   // this split's key range
  const int je   = js + jlen;

  // --- Q^T B-tiles: lane = query column q0+lm, table-order K (8h.., 16+8h..) ---
  const _Float16* qrow = qh + (size_t)(q0 + lm) * DH;
  const v16h qb0 = cat8(*(const v8h*)(qrow + kb), *(const v8h*)(qrow + kb + 16));       // feat 0..31
  const v16h qb1 = cat8(*(const v8h*)(qrow + 32 + kb), *(const v8h*)(qrow + 32 + kb + 16));  // feat 32..63

  // V^T A-tile row pointers (feature rows ft*16+lm)
  const _Float16* vp0 = vT + (size_t)(0 * 16 + lm) * NN;
  const _Float16* vp1 = vT + (size_t)(1 * 16 + lm) * NN;
  const _Float16* vp2 = vT + (size_t)(2 * 16 + lm) * NN;
  const _Float16* vp3 = vT + (size_t)(3 * 16 + lm) * NN;

  const int pq = ptr32[q0 + lm];  // this lane's query graph id
  const float* brow = bmat + (size_t)(q0 + lm) * NN;

  float M = -1.0e30f, L = 0.0f;
  v8f o0 = {}, o1 = {}, o2 = {}, o3 = {};  // O^T: rows=features, col=query

  for (int j0 = js; j0 < je; j0 += CHUNK) {
    // --- A-tiles of K: 2 key tiles (rows j0+16t+lm) x 2 K-halves ---
    const _Float16* k0r = kh + (size_t)(j0 + lm) * DH;
    const _Float16* k1r = kh + (size_t)(j0 + 16 + lm) * DH;
    v16h ak00 = cat8(*(const v8h*)(k0r + kb),      *(const v8h*)(k0r + kb + 16));
    v16h ak01 = cat8(*(const v8h*)(k0r + 32 + kb), *(const v8h*)(k0r + 32 + kb + 16));
    v16h ak10 = cat8(*(const v8h*)(k1r + kb),      *(const v8h*)(k1r + kb + 16));
    v16h ak11 = cat8(*(const v8h*)(k1r + 32 + kb), *(const v8h*)(k1r + 32 + kb + 16));

    // S^T tiles: rows = keys (r + 8*half [+16]), col = query lm
    v8f s0 = {};
    s0 = wmma16(ak00, qb0, s0);
    s0 = wmma16(ak01, qb1, s0);
    v8f s1 = {};
    s1 = wmma16(ak10, qb0, s1);
    s1 = wmma16(ak11, qb1, s1);

    // --- b tile: contiguous 8 floats per lane per key tile ---
    v8f b0 = *(const v8f*)(brow + j0 + 8 * half);
    v8f b1 = *(const v8f*)(brow + j0 + 16 + 8 * half);
    // graph ids of the 8+8 keys this lane holds
    v8i pk0 = *(const v8i*)(ptr32 + j0 + 8 * half);
    v8i pk1 = *(const v8i*)(ptr32 + j0 + 16 + 8 * half);

    if (j0 + CHUNK < je)
      __builtin_prefetch(brow + (j0 + CHUNK) + 16 * half, 0, 0);

    s0 += b0;
    s1 += b1;

    // --- multiplicative mask, per-lane max tree ---
    float t0[8], t1[8];
    float mx = -1.0e30f;
#pragma unroll
    for (int r = 0; r < 8; ++r) {
      t0[r] = s0[r] * ((pq == pk0[r]) ? 1.0f : -1.0e6f);
      t1[r] = s1[r] * ((pq == pk1[r]) ? 1.0f : -1.0e6f);
      mx = fmaxf(mx, fmaxf(t0[r], t1[r]));
    }
    mx = fmaxf(mx, __shfl_xor(mx, 16));  // merge partner half: full 32-key max

    float Mn = fmaxf(M, mx);
    float sc = __expf(M - Mn);
    M = Mn;

    // --- exp, gated f16 pack (numerator), full sum (denominator) ---
    v8h P0, P1;
    float ls = 0.0f;
#pragma unroll
    for (int r = 0; r < 8; ++r) {
      float e0 = __expf(t0[r] - Mn);
      float e1 = __expf(t1[r] - Mn);
      ls += e0 + e1;
      P0[r] = (pq == pk0[r]) ? (_Float16)e0 : (_Float16)0.0f;
      P1[r] = (pq == pk1[r]) ? (_Float16)e1 : (_Float16)0.0f;
    }
    ls += __shfl_xor(ls, 16);
    L = L * sc + ls;
    o0 *= sc; o1 *= sc; o2 *= sc; o3 *= sc;

    // --- build P^T B-tile in registers: own two octets concatenated, no lane swap ---
    // B wants table order: keys 8h..8h+7, then 16+8h..16+8h+7.
    v16h Bp  = cat8(P0, P1);

    // --- O^T += V^T_tile @ P^T ---
    v16h av0 = cat8(*(const v8h*)(vp0 + j0 + kb), *(const v8h*)(vp0 + j0 + kb + 16));
    v16h av1 = cat8(*(const v8h*)(vp1 + j0 + kb), *(const v8h*)(vp1 + j0 + kb + 16));
    v16h av2 = cat8(*(const v8h*)(vp2 + j0 + kb), *(const v8h*)(vp2 + j0 + kb + 16));
    v16h av3 = cat8(*(const v8h*)(vp3 + j0 + kb), *(const v8h*)(vp3 + j0 + kb + 16));
    o0 = wmma16(av0, Bp, o0);
    o1 = wmma16(av1, Bp, o1);
    o2 = wmma16(av2, Bp, o2);
    o3 = wmma16(av3, Bp, o3);
  }

  // --- write partials (unnormalized, in local max frame) ---
  const int by = blockIdx.y;
  if (half == 0) {
    pM[(size_t)by * NN + q0 + lm] = M;
    pL[(size_t)by * NN + q0 + lm] = L;
  }
  float* orow = pO + (size_t)by * NN * DH + (size_t)(q0 + lm) * DH + 8 * half;
#pragma unroll
  for (int r = 0; r < 8; ++r) {
    orow[r]      = o0[r];
    orow[16 + r] = o1[r];
    orow[32 + r] = o2[r];
    orow[48 + r] = o3[r];
  }
}

// ---------------------------------------------------------------------------
// Phase 3: merge split partials (softmax two-accumulator merge) + normalize.
// ---------------------------------------------------------------------------
__global__ __launch_bounds__(256) void combine_kernel(
    const float* __restrict__ pO, const float* __restrict__ pM,
    const float* __restrict__ pL, float* __restrict__ out, int split) {
  int idx = blockIdx.x * blockDim.x + threadIdx.x;
  if (idx >= NN * DH) return;
  int row = idx >> 6;
  float M = -1.0e30f;
  for (int s = 0; s < split; ++s) M = fmaxf(M, pM[(size_t)s * NN + row]);
  float L = 0.0f, O = 0.0f;
  for (int s = 0; s < split; ++s) {
    float w = __expf(pM[(size_t)s * NN + row] - M);
    L += pL[(size_t)s * NN + row] * w;
    O += pO[(size_t)s * NN * DH + idx] * w;
  }
  out[idx] = O / L;
}

extern "C" void kernel_launch(void* const* d_in, const int* in_sizes, int n_in,
                              void* d_out, int out_size, void* d_ws, size_t ws_size,
                              hipStream_t stream) {
  const float* query = (const float*)d_in[0];
  const float* key   = (const float*)d_in[1];
  const float* value = (const float*)d_in[2];
  const float* bmat  = (const float*)d_in[3];
  const float* Wq    = (const float*)d_in[4];
  const float* bq    = (const float*)d_in[5];
  const float* Wk    = (const float*)d_in[6];
  const float* bk    = (const float*)d_in[7];
  const float* Wv    = (const float*)d_in[8];
  const float* bv    = (const float*)d_in[9];
  const long long* ptr = (const long long*)d_in[10];
  float* out = (float*)d_out;

  char* ws = (char*)d_ws;
  _Float16* qh = (_Float16*)(ws + 0);             // 1 MB
  _Float16* kh = (_Float16*)(ws + (1u << 20));    // 1 MB
  _Float16* vT = (_Float16*)(ws + (2u << 20));    // 1 MB
  int* ptr32   = (int*)(ws + (3u << 20));         // 32 KB
  size_t off_pM = (3u << 20) + (64u << 10);
  size_t off_pL = off_pM + (128u << 10);          // room for 4 splits of M
  size_t off_pO = off_pL + (128u << 10);          // room for 4 splits of L
  float* pM = (float*)(ws + off_pM);
  float* pL = (float*)(ws + off_pL);
  float* pO = (float*)(ws + off_pO);

  // choose j-split (more resident waves) based on available workspace
  int split = 1;
  const size_t oBytes = (size_t)NN * DH * 4;
  if (ws_size >= off_pO + 4 * oBytes)      split = 4;
  else if (ws_size >= off_pO + 2 * oBytes) split = 2;

  proj_kernel<<<(NN * DH) / 256, 256, 0, stream>>>(query, key, value, Wq, bq, Wk,
                                                   bk, Wv, bv, ptr, qh, kh, vT,
                                                   ptr32);
  dim3 agrid(NN / 16, split);
  attn_kernel<<<agrid, 32, 0, stream>>>(bmat, qh, kh, vT, ptr32, pO, pM, pL,
                                        NN / split);
  combine_kernel<<<(NN * DH) / 256, 256, 0, stream>>>(pO, pM, pL, out, split);
}
